// MultiHeadSelfAttention_23811298689031
// MI455X (gfx1250) — compile-verified, matrix operand order checked by reading, not run on hardware
//
#include <hip/hip_runtime.h>
#include <hip/hip_bf16.h>

// ---- types -----------------------------------------------------------------
typedef __bf16 v16bf __attribute__((ext_vector_type(16)));
typedef __bf16 bf8   __attribute__((ext_vector_type(8)));
typedef float  v8f   __attribute__((ext_vector_type(8)));
typedef unsigned v4u __attribute__((ext_vector_type(4)));
typedef int      v8i __attribute__((ext_vector_type(8)));
typedef int      v4i __attribute__((ext_vector_type(4)));

#define D_MODEL 1024
#define SEQ     2048
#define BATCH   2
#define HEADS   16
#define HD      64
#define TOK     (BATCH * SEQ)   // 4096
#define KTILE   64              // attention key tile

// ---- helpers ---------------------------------------------------------------
__device__ __forceinline__ __bf16 f2bf(float x) {
  unsigned u = __builtin_bit_cast(unsigned, x);
  unsigned r = (u + 0x7FFFu + ((u >> 16) & 1u)) >> 16;   // RNE
  unsigned short h = (unsigned short)r;
  return __builtin_bit_cast(__bf16, h);
}

// packed f32x2 -> bf16x2 (RNE) in one VALU op
__device__ __forceinline__ unsigned cvt_pk_bf16(float lo, float hi) {
  unsigned r;
  asm("v_cvt_pk_bf16_f32 %0, %1, %2" : "=v"(r) : "v"(lo), "v"(hi));
  return r;
}
__device__ __forceinline__ __bf16 bf_lo(unsigned v) {
  return __builtin_bit_cast(__bf16, (unsigned short)v);
}
__device__ __forceinline__ __bf16 bf_hi(unsigned v) {
  return __builtin_bit_cast(__bf16, (unsigned short)(v >> 16));
}

__device__ __forceinline__ v8f wmma_bf16(v16bf a, v16bf b, v8f c) {
  return __builtin_amdgcn_wmma_f32_16x16x32_bf16(
      false, a, false, b, (short)0, c, false, false);
}

// Load a 16x32 A-style bf16 fragment from row-major source (row stride = ld).
__device__ __forceinline__ v16bf load_frag(const __bf16* src, int ld) {
  const int lane = threadIdx.x & 31;
  const __bf16* p = src + (lane & 15) * ld + ((lane & 16) >> 1);
  bf8 lo = *(const bf8*)p;
  bf8 hi = *(const bf8*)(p + 16);
  return __builtin_shufflevector(lo, hi,
      0, 1, 2, 3, 4, 5, 6, 7, 8, 9, 10, 11, 12, 13, 14, 15);
}

// Async 16B/lane global->LDS copy (GVS), tracked by ASYNCcnt.
__device__ __forceinline__ void async_g2l_b128(const void* lds_ptr,
                                               const void* gbase, int byte_off) {
  unsigned lds = (unsigned)(uintptr_t)lds_ptr;
  asm volatile("global_load_async_to_lds_b128 %0, %1, %2"
               :: "v"(lds), "v"(byte_off), "s"(gbase) : "memory");
}
__device__ __forceinline__ void wait_async0() {
  asm volatile("s_wait_asynccnt 0x0" ::: "memory");
}

// TDM: 2D (or 1D when tile_d1==0) tensor tile -> LDS, bf16 elements.
// D# layout per CDNA5 ISA; groups 2-3 zero (<=2D).
__device__ __forceinline__ void tdm_load_2d(unsigned lds_off, const void* gaddr,
                                            unsigned tensor_d0, unsigned tensor_d1,
                                            unsigned tile_d0, unsigned tile_d1,
                                            unsigned stride0) {
  unsigned long long ga = (unsigned long long)(uintptr_t)gaddr;
  v4u g0;
  g0[0] = 1u;                                   // count=1, user mode
  g0[1] = lds_off;                              // lds_addr
  g0[2] = (unsigned)ga;                         // global_addr[31:0]
  g0[3] = (unsigned)(ga >> 32) | (2u << 30);    // global_addr[56:32] | type=2
  v8i g1;
  g1[0] = 1 << 16;                              // data_size=1 (2 bytes)
  g1[1] = (int)(tensor_d0 << 16);               // tensor_dim0[15:0]
  g1[2] = (int)((tensor_d0 >> 16) | (tensor_d1 << 16));
  g1[3] = (int)((tensor_d1 >> 16) | (tile_d0 << 16));
  g1[4] = (int)tile_d1;                         // tile_dim1 | tile_dim2=0
  g1[5] = (int)stride0;                         // tensor_dim0_stride[31:0]
  g1[6] = 0;
  g1[7] = 0;
  v4i zero4 = {0, 0, 0, 0};
  v8i zero8 = {0, 0, 0, 0, 0, 0, 0, 0};
  __builtin_amdgcn_tensor_load_to_lds(g0, g1, zero4, zero4, zero8, 0);
}

// ---- DPP cross-lane reductions within each 16-lane half (VALU only) --------
constexpr int dpp8_sel_xor(int m) {
  int v = 0;
  for (int j = 0; j < 8; ++j) v |= ((j ^ m) << (3 * j));
  return v;
}
template <int SEL>
__device__ __forceinline__ float dpp8_f(float x) {
  return __builtin_bit_cast(float,
      __builtin_amdgcn_mov_dpp8(__builtin_bit_cast(int, x), SEL));
}
__device__ __forceinline__ float dpp_ror8(float x) {
  return __builtin_bit_cast(float,
      __builtin_amdgcn_mov_dpp(__builtin_bit_cast(int, x), 0x128, 0xf, 0xf, true));
}
__device__ __forceinline__ float max_red16(float x) {
  x = fmaxf(x, dpp8_f<dpp8_sel_xor(1)>(x));
  x = fmaxf(x, dpp8_f<dpp8_sel_xor(2)>(x));
  x = fmaxf(x, dpp8_f<dpp8_sel_xor(4)>(x));
  x = fmaxf(x, dpp_ror8(x));
  return x;
}
__device__ __forceinline__ float sum_red16(float x) {
  x += dpp8_f<dpp8_sel_xor(1)>(x);
  x += dpp8_f<dpp8_sel_xor(2)>(x);
  x += dpp8_f<dpp8_sel_xor(4)>(x);
  x += dpp_ror8(x);
  return x;
}

// ---- fp32 -> bf16 conversion (vectorized b128 in / b128 out) ---------------
__global__ void cvt_kernel(const float* __restrict__ src,
                           __bf16* __restrict__ dst, int n) {
  int i = (blockIdx.x * blockDim.x + threadIdx.x) * 8;
  int stride = gridDim.x * blockDim.x * 8;
  for (; i < n; i += stride) {
    float4 a = *(const float4*)(src + i);
    float4 b = *(const float4*)(src + i + 4);
    uint4 o;
    o.x = cvt_pk_bf16(a.x, a.y);
    o.y = cvt_pk_bf16(a.z, a.w);
    o.z = cvt_pk_bf16(b.x, b.y);
    o.w = cvt_pk_bf16(b.z, b.w);
    *(uint4*)(dst + i) = o;
  }
}

// ---- fp32 -> bf16 conversion + 1024x1024 transpose: dst[n][k] = src[k][n] --
__global__ __launch_bounds__(256) void cvt_t_kernel(const float* __restrict__ src,
                                                    __bf16* __restrict__ dst) {
  __shared__ __bf16 tile[32][33];
  const int tx = threadIdx.x & 31;
  const int ty = threadIdx.x >> 5;
  const int c0 = blockIdx.x * 32, r0 = blockIdx.y * 32;
#pragma unroll
  for (int p = 0; p < 4; ++p) {
    int r = ty + p * 8;
    tile[r][tx] = f2bf(src[(size_t)(r0 + r) * D_MODEL + c0 + tx]);
  }
  __syncthreads();
#pragma unroll
  for (int p = 0; p < 4; ++p) {
    int rr = ty + p * 8;
    dst[(size_t)(c0 + rr) * D_MODEL + r0 + tx] = tile[tx][rr];
  }
}

// ---- WMMA GEMM: [4096,1024] x Wt[1024 n][1024 k], K-step 64, async 2-buf ---
// MODE 0: bf16 out [B*H][SEQ][HD]   (Q, K)
// MODE 1: bf16 out [B*H][HD][SEQ]   (V, pre-transposed for attention staging)
// MODE 2: f32  out [TOK][D_MODEL]   (final projection)
template <int MODE>
__global__ __launch_bounds__(256) void gemm_kernel(
    const __bf16* __restrict__ A,     // [TOK][D_MODEL]
    const __bf16* __restrict__ Wt,    // [n][k] (transposed W)
    const float* __restrict__ bias,
    void* __restrict__ outv) {
  __shared__ __bf16 As[2][128 * 64];  // [m][k]
  __shared__ __bf16 Bs[2][128 * 64];  // [n][k]
  const int tid = threadIdx.x;
  const int lane = tid & 31, wave = tid >> 5;
  const int wm = wave >> 2, wn = wave & 3;
  const int m0 = blockIdx.y * 128, n0 = blockIdx.x * 128;
  v8f z = {};
  v8f acc[4][2];
#pragma unroll
  for (int i = 0; i < 4; ++i)
#pragma unroll
    for (int j = 0; j < 2; ++j) acc[i][j] = z;

#define STAGE_AB(BUF, K0)                                                     \
  {                                                                           \
    _Pragma("unroll") for (int p = 0; p < 4; ++p) {                           \
      int idx = (p * 256 + tid) * 8;                                          \
      int rr = idx >> 6, cc = idx & 63;                                       \
      async_g2l_b128(As[BUF] + idx, A,  ((m0 + rr) * D_MODEL + (K0) + cc) * 2); \
      async_g2l_b128(Bs[BUF] + idx, Wt, ((n0 + rr) * D_MODEL + (K0) + cc) * 2); \
    }                                                                         \
  }

  STAGE_AB(0, 0);
  int buf = 0;
  for (int k0 = 0; k0 < D_MODEL; k0 += 64, buf ^= 1) {
    wait_async0();
    __syncthreads();
    if (k0 + 64 < D_MODEL) STAGE_AB(buf ^ 1, k0 + 64);
#pragma unroll
    for (int kh = 0; kh < 2; ++kh) {
      v16bf af[4], bfr[2];
#pragma unroll
      for (int i = 0; i < 4; ++i)
        af[i] = load_frag(As[buf] + (wm * 64 + i * 16) * 64 + kh * 32, 64);
#pragma unroll
      for (int j = 0; j < 2; ++j)
        bfr[j] = load_frag(Bs[buf] + (wn * 32 + j * 16) * 64 + kh * 32, 64);
#pragma unroll
      for (int i = 0; i < 4; ++i)
#pragma unroll
        for (int j = 0; j < 2; ++j)
          acc[i][j] = wmma_bf16(af[i], bfr[j], acc[i][j]);
    }
  }
#undef STAGE_AB

  const int cl = lane & 15, hb = (lane >> 4) * 8;
#pragma unroll
  for (int i = 0; i < 4; ++i) {
#pragma unroll
    for (int j = 0; j < 2; ++j) {
      int c = n0 + wn * 32 + j * 16 + cl;
      float bv = bias[c];
      if (MODE == 2) {
        float* out = (float*)outv;
#pragma unroll
        for (int r = 0; r < 8; ++r) {
          int t = m0 + wm * 64 + i * 16 + r + hb;
          out[(size_t)t * D_MODEL + c] = acc[i][j][r] + bv;
        }
      } else {
        __bf16* out = (__bf16*)outv;
        int h = c >> 6, dd = c & 63;
#pragma unroll
        for (int r = 0; r < 8; r += 2) {
          int t = m0 + wm * 64 + i * 16 + r + hb;
          int b = t >> 11, s = t & (SEQ - 1);
          unsigned pk = cvt_pk_bf16(acc[i][j][r] + bv, acc[i][j][r + 1] + bv);
          if (MODE == 0) {
            size_t base = (((size_t)(b * HEADS + h) * SEQ) + s) * HD + dd;
            out[base] = bf_lo(pk);
            out[base + HD] = bf_hi(pk);
          } else {  // MODE 1: [bh][dd][s], s/s+1 contiguous -> b32 store
            size_t base = (((size_t)(b * HEADS + h) * HD) + dd) * SEQ + s;
            *(unsigned*)(out + base) = pk;
          }
        }
      }
    }
  }
}

// ---- flash attention: one WG per (b,h) x 128-query band, 64-key tiles ------
// K/V tiles staged by the Tensor Data Mover (one descriptor pair per tile,
// issued by wave 0, TENSORcnt + barrier for visibility, double-buffered).
__global__ __launch_bounds__(256) void attn_kernel(
    const __bf16* __restrict__ Q,     // [B*H][SEQ][HD]
    const __bf16* __restrict__ K,     // [B*H][SEQ][HD]
    const __bf16* __restrict__ Vt,    // [B*H][HD][SEQ]
    __bf16* __restrict__ C) {         // [TOK][D_MODEL]
  __shared__ __bf16 Ks[2][KTILE * HD];   // [key][dim]  8KB x2
  __shared__ __bf16 Vs[2][HD * KTILE];   // [dim][key]  8KB x2
  __shared__ __bf16 Ps[8][16 * KTILE];   // per-wave [qrow][key] 16KB
  const int tid = threadIdx.x, lane = tid & 31, wave = tid >> 5;
  const int bh = blockIdx.y;
  const int b = bh >> 4, h = bh & 15;
  const int q0 = blockIdx.x * 128;
  const __bf16* Qh = Q  + (size_t)bh * SEQ * HD;
  const __bf16* Kh = K  + (size_t)bh * SEQ * HD;
  const __bf16* Vh = Vt + (size_t)bh * HD * SEQ;
  const int qrow = q0 + wave * 16;

  v16bf qf0 = load_frag(Qh + (size_t)qrow * HD, HD);
  v16bf qf1 = load_frag(Qh + (size_t)qrow * HD + 32, HD);

  v8f z = {};
  v8f o[4];
#pragma unroll
  for (int j = 0; j < 4; ++j) o[j] = z;
  float m[8], l[8];
#pragma unroll
  for (int r = 0; r < 8; ++r) { m[r] = -1e30f; l[r] = 0.f; }
  const float scale = 0.125f;          // 1/sqrt(64)
  __bf16* Pw = Ps[wave];
  const int cl = lane & 15, hb = (lane >> 4) * 8;

  // K tile: 64 contiguous rows x 64 dims = contiguous 8KB -> 1D descriptor.
  // V tile: 64 dims x 64 keys from [HD][SEQ]     -> 2D descriptor, stride SEQ.
#define STAGE_KV_TDM(BUF, KT0)                                                \
  {                                                                           \
    tdm_load_2d((unsigned)(uintptr_t)(void*)Ks[BUF], Kh + (size_t)(KT0) * HD, \
                SEQ * HD, 1, KTILE * HD, 0, SEQ * HD);                        \
    tdm_load_2d((unsigned)(uintptr_t)(void*)Vs[BUF], Vh + (KT0),              \
                SEQ, HD, KTILE, HD, SEQ);                                     \
  }

  if (tid < 32) STAGE_KV_TDM(0, 0);
  int buf = 0;
  for (int kt = 0; kt < SEQ; kt += KTILE, buf ^= 1) {
    if (tid < 32) __builtin_amdgcn_s_wait_tensorcnt(0);
    __syncthreads();
    if (kt + KTILE < SEQ && tid < 32) STAGE_KV_TDM(buf ^ 1, kt + KTILE);
    const __bf16* Kb = Ks[buf];
    const __bf16* Vb = Vs[buf];

    // S = Q K^T : 16 q-rows x 64 keys (4 n-subtiles x 2 k-halves)
    v8f sacc[4];
#pragma unroll
    for (int ns = 0; ns < 4; ++ns) {
      v16bf kf0 = load_frag(Kb + ns * 16 * HD, HD);
      v16bf kf1 = load_frag(Kb + ns * 16 * HD + 32, HD);
      v8f s = wmma_bf16(qf0, kf0, z);
      sacc[ns] = wmma_bf16(qf1, kf1, s);
    }

    // online softmax; reductions on VALU via DPP
    float alpha[8];
#pragma unroll
    for (int r = 0; r < 8; ++r) {
      float x0 = sacc[0][r] * scale, x1 = sacc[1][r] * scale;
      float x2 = sacc[2][r] * scale, x3 = sacc[3][r] * scale;
      float mx = max_red16(fmaxf(fmaxf(x0, x1), fmaxf(x2, x3)));
      float mn = fmaxf(m[r], mx);
      float p0 = __expf(x0 - mn), p1 = __expf(x1 - mn);
      float p2 = __expf(x2 - mn), p3 = __expf(x3 - mn);
      float rs = sum_red16((p0 + p1) + (p2 + p3));
      alpha[r] = __expf(m[r] - mn);
      l[r] = l[r] * alpha[r] + rs;
      m[r] = mn;
      unsigned pkA = cvt_pk_bf16(p0, p1);
      unsigned pkB = cvt_pk_bf16(p2, p3);
      __bf16* pr = Pw + (r + hb) * KTILE + cl;
      pr[0]  = bf_lo(pkA);
      pr[16] = bf_hi(pkA);
      pr[32] = bf_lo(pkB);
      pr[48] = bf_hi(pkB);
    }
#pragma unroll
    for (int j = 0; j < 4; ++j)
#pragma unroll
      for (int r = 0; r < 8; ++r) o[j][r] *= alpha[r];

    // O += P V : K dim = 64 keys (2 halves), N = 64 dims (4 chunks)
    v16bf pf0 = load_frag(Pw, KTILE);
    v16bf pf1 = load_frag(Pw + 32, KTILE);
#pragma unroll
    for (int j = 0; j < 4; ++j) {
      v16bf vf0 = load_frag(Vb + j * 16 * KTILE, KTILE);
      v16bf vf1 = load_frag(Vb + j * 16 * KTILE + 32, KTILE);
      o[j] = wmma_bf16(pf0, vf0, o[j]);
      o[j] = wmma_bf16(pf1, vf1, o[j]);
    }
  }
#undef STAGE_KV_TDM

  // epilogue: O / l -> concat[B,S,D] (bf16), packed pair conversion
  float inv[8];
#pragma unroll
  for (int r = 0; r < 8; ++r) inv[r] = 1.f / l[r];
#pragma unroll
  for (int j = 0; j < 4; ++j) {
#pragma unroll
    for (int r = 0; r < 8; r += 2) {
      unsigned pk = cvt_pk_bf16(o[j][r] * inv[r], o[j][r + 1] * inv[r + 1]);
      int col = h * 64 + j * 16 + cl;
      size_t base = ((size_t)(b * SEQ + qrow + r + hb)) * D_MODEL + col;
      C[base] = bf_lo(pk);
      C[base + D_MODEL] = bf_hi(pk);
    }
  }
}

// ---- launcher --------------------------------------------------------------
extern "C" void kernel_launch(void* const* d_in, const int* in_sizes, int n_in,
                              void* d_out, int out_size, void* d_ws, size_t ws_size,
                              hipStream_t stream) {
  const float* X  = (const float*)d_in[0];
  const float* Wq = (const float*)d_in[1];
  const float* bq = (const float*)d_in[2];
  const float* Wk = (const float*)d_in[3];
  const float* bk = (const float*)d_in[4];
  const float* Wv = (const float*)d_in[5];
  const float* bv = (const float*)d_in[6];
  const float* Wo = (const float*)d_in[7];
  const float* bo = (const float*)d_in[8];

  char* ws = (char*)d_ws;
  const size_t MB = 1024ull * 1024ull;
  __bf16* Xb  = (__bf16*)(ws + 0);        //  8 MB: X bf16 [4096][1024]
  __bf16* Wqt = (__bf16*)(ws + 8  * MB);  //  2 MB each, transposed [n][k]
  __bf16* Wkt = (__bf16*)(ws + 10 * MB);
  __bf16* Wvt = (__bf16*)(ws + 12 * MB);
  __bf16* Wot = (__bf16*)(ws + 14 * MB);
  __bf16* Qb  = (__bf16*)(ws + 16 * MB);  //  8 MB, [B*H][SEQ][HD]
  __bf16* Kb  = (__bf16*)(ws + 24 * MB);  //  8 MB, [B*H][SEQ][HD]
  __bf16* Vb  = (__bf16*)(ws + 32 * MB);  //  8 MB, [B*H][HD][SEQ]
  __bf16* Cb  = (__bf16*)(ws + 40 * MB);  //  8 MB concat bf16 [4096][1024]

  cvt_kernel<<<2048, 256, 0, stream>>>(X, Xb, TOK * D_MODEL);
  dim3 gt(32, 32);
  cvt_t_kernel<<<gt, 256, 0, stream>>>(Wq, Wqt);
  cvt_t_kernel<<<gt, 256, 0, stream>>>(Wk, Wkt);
  cvt_t_kernel<<<gt, 256, 0, stream>>>(Wv, Wvt);
  cvt_t_kernel<<<gt, 256, 0, stream>>>(Wo, Wot);

  dim3 g(D_MODEL / 128, TOK / 128);       // (8, 32)
  gemm_kernel<0><<<g, 256, 0, stream>>>(Xb, Wqt, bq, Qb);
  gemm_kernel<0><<<g, 256, 0, stream>>>(Xb, Wkt, bk, Kb);
  gemm_kernel<1><<<g, 256, 0, stream>>>(Xb, Wvt, bv, Vb);

  attn_kernel<<<dim3(SEQ / 128, BATCH * HEADS), 256, 0, stream>>>(Qb, Kb, Vb, Cb);

  gemm_kernel<2><<<g, 256, 0, stream>>>(Cb, Wot, bo, (float*)d_out);
}
